// StructureLayer_12618613915751
// MI455X (gfx1250) — matrix operand order repaired and checked by reading, not compiled, not run on hardware
//
#include <hip/hip_runtime.h>
#include <math.h>

typedef __attribute__((ext_vector_type(16))) _Float16 v16h;
typedef __attribute__((ext_vector_type(8)))  _Float16 v8h;
typedef __attribute__((ext_vector_type(8)))  float    v8f;
typedef __attribute__((__vector_size__(16))) int      i32x4;

#define C_Sd 384
#define C_Pd 128
#define C_Hd 16
#define Hh   12
#define Nn   512
#define CTOT 1152   /* 192 q + 384 kv + 144 qp + 144 kp + 288 vp */
#define FEAT 2112   /* 12*(16 + 24 + 8 + 128) */

#define AS1 __attribute__((address_space(1)))
#define AS3 __attribute__((address_space(3)))

#if defined(__has_builtin)
#  if __has_builtin(__builtin_amdgcn_global_load_async_to_lds_b128) && \
      __has_builtin(__builtin_amdgcn_s_wait_asynccnt)
#    define USE_ASYNC_LDS 1
#  endif
#endif
#ifndef USE_ASYNC_LDS
#  define USE_ASYNC_LDS 0
#endif

// ---------------- WMMA fragment loaders (16-bit, contiguous k = 16h + j, same for A and B) ----------------
// A (16x32 f16): lane m=lane&15, half=lane>>4; elems j: K=16h+j
__device__ __forceinline__ v16h ld_a16(const _Float16* rowk0, int half) {
  v8h lo = *(const v8h*)(rowk0 + 16 * half);
  v8h hi = *(const v8h*)(rowk0 + 16 * half + 8);
  v16h r;
#pragma unroll
  for (int j = 0; j < 8; ++j) { r[j] = lo[j]; r[8 + j] = hi[j]; }
  return r;
}
// B (32x16 f16, column per lane): lane n=lane&15, half=lane>>4; elems j: K=16h+j
__device__ __forceinline__ v16h ld_b16(const _Float16* colk0, int half) {
  v8h lo = *(const v8h*)(colk0 + 16 * half);
  v8h hi = *(const v8h*)(colk0 + 16 * half + 8);
  v16h r;
#pragma unroll
  for (int j = 0; j < 8; ++j) { r[j] = lo[j]; r[8 + j] = hi[j]; }
  return r;
}

// ---------------- generic WMMA GEMM: C[M][N] = A[M][K] * Bt[N][K]^T (+bias,+res,relu) ----
__global__ void gemm_wmma(const _Float16* __restrict__ A, const _Float16* __restrict__ Bt,
                          const float* __restrict__ bias, const float* __restrict__ res,
                          float* __restrict__ C, _Float16* __restrict__ Ch,
                          int M, int N, int K, int relu)
{
  const int lane = threadIdx.x & 31, wv = threadIdx.x >> 5;
  const int tilesN = N >> 4;
  const int tile = blockIdx.x * 4 + wv;
  if (tile >= (M >> 4) * tilesN) return;
  const int m0 = (tile / tilesN) << 4, n0 = (tile % tilesN) << 4;
  const int half = lane >> 4, mr = lane & 15;
  const _Float16* arow = A + (size_t)(m0 + mr) * K;
  const _Float16* brow = Bt + (size_t)(n0 + mr) * K;
  v8f acc = {};
  for (int k0 = 0; k0 < K; k0 += 32) {
    v16h a = ld_a16(arow + k0, half);
    v16h b = ld_b16(brow + k0, half);
    acc = __builtin_amdgcn_wmma_f32_16x16x32_f16(false, a, false, b, (short)0, acc, false, false);
  }
  const int n = n0 + mr;
  const float bv = bias ? bias[n] : 0.f;
#pragma unroll
  for (int r = 0; r < 8; ++r) {
    int m = m0 + r + 8 * half;
    float v = acc[r] + bv;
    if (res) v += res[(size_t)m * N + n];
    if (relu && v < 0.f) v = 0.f;
    C[(size_t)m * N + n] = v;
    if (Ch) Ch[(size_t)m * N + n] = (_Float16)v;
  }
}

// ---------------- z = sqrt(1/3)*(p @ b_w + b_b); p converted to f16 on the fly ---------
__global__ void pair_bias_gemm(const float* __restrict__ p, const _Float16* __restrict__ Btb,
                               const float* __restrict__ b_b, float* __restrict__ z)
{
  const int lane = threadIdx.x & 31, wv = threadIdx.x >> 5;
  const int tile = blockIdx.x * 4 + wv;           // 16384 row tiles of p (M = 512*512)
  const int m0 = tile << 4;
  const int half = lane >> 4, mr = lane & 15;
  const float* arow = p + (size_t)(m0 + mr) * C_Pd;
  const _Float16* brow = Btb + mr * C_Pd;
  v8f acc = {};
  for (int k0 = 0; k0 < C_Pd; k0 += 32) {
    v16h a;
#pragma unroll
    for (int j = 0; j < 8; ++j) {
      a[j]     = (_Float16)arow[k0 + 16 * half + j];
      a[8 + j] = (_Float16)arow[k0 + 16 * half + 8 + j];
    }
    v16h b = ld_b16(brow + k0, half);
    acc = __builtin_amdgcn_wmma_f32_16x16x32_f16(false, a, false, b, (short)0, acc, false, false);
  }
  const int n = mr;
  const float s3 = 0.57735026918962576f;
  if (n < Hh) {
    const float bb = b_b[n];
#pragma unroll
    for (int r = 0; r < 8; ++r) {
      int m = m0 + r + 8 * half;
      z[(size_t)m * Hh + n] = (acc[r] + bb) * s3;
    }
  }
}

// ---------------- rigid transform of projected points ---------------------------------
__global__ void to_pts(const float* __restrict__ proj, int colbase, int npts,
                       const float* __restrict__ rot, const float* __restrict__ trans,
                       float* __restrict__ dst)
{
  int t = blockIdx.x * blockDim.x + threadIdx.x;
  if (t >= Nn * npts) return;
  int i = t / npts, idx = t % npts;
  const float* pr = proj + (size_t)i * CTOT + colbase + idx * 3;
  float lx = pr[0], ly = pr[1], lz = pr[2];
  const float* R = rot + i * 9; const float* tr = trans + i * 3;
  float* d = dst + (size_t)i * (npts * 3) + idx * 3;
  d[0] = R[0] * lx + R[1] * ly + R[2] * lz + tr[0];
  d[1] = R[3] * lx + R[4] * ly + R[5] * lz + tr[1];
  d[2] = R[6] * lx + R[7] * ly + R[8] * lz + tr[2];
}

// ---------------- build augmented Q/K (K-dim 32): scalar qk + point-distance terms ----
__global__ void prep_qk(const float* __restrict__ proj, const float* __restrict__ qpts,
                        const float* __restrict__ kpts, const float* __restrict__ head_w,
                        _Float16* __restrict__ Qa, _Float16* __restrict__ Ka)
{
  int t = blockIdx.x * blockDim.x + threadIdx.x;
  if (t >= Hh * Nn) return;
  int h = t / Nn, i = t % Nn;
  const float qk_scale = 0.14433756729740643f;                  // sqrt(1/(3*16))
  float hw = logf(1.f + expf(head_w[h])) * 0.13608276348795434f; // softplus * sqrt(1/54)
  _Float16* qo = Qa + ((size_t)h * Nn + i) * 32;
  _Float16* ko = Ka + ((size_t)h * Nn + i) * 32;
  const float* q = proj + (size_t)i * CTOT + h * C_Hd;
  const float* k = proj + (size_t)i * CTOT + 192 + h * 32;
#pragma unroll
  for (int c = 0; c < 16; ++c) { qo[c] = (_Float16)(qk_scale * q[c]); ko[c] = (_Float16)k[c]; }
  const float* qp = qpts + (size_t)i * 144 + h * 12;
  const float* kp = kpts + (size_t)i * 144 + h * 12;
  float qn = 0.f, kn = 0.f;
#pragma unroll
  for (int d = 0; d < 12; ++d) {
    qo[16 + d] = (_Float16)(hw * qp[d]); ko[16 + d] = (_Float16)kp[d];
    qn += qp[d] * qp[d]; kn += kp[d] * kp[d];
  }
  qo[28] = (_Float16)(-0.5f * hw * qn); ko[28] = (_Float16)1.f;
  qo[29] = (_Float16)1.f;               ko[29] = (_Float16)(-0.5f * hw * kn);
  qo[30] = (_Float16)0.f; ko[30] = (_Float16)0.f;
  qo[31] = (_Float16)0.f; ko[31] = (_Float16)0.f;
}

// ---------------- attention logits: one WMMA (K=32) per 16x16 tile + bias + mask ------
__global__ void attn_logits(const _Float16* __restrict__ Qa, const _Float16* __restrict__ Ka,
                            const float* __restrict__ z, const float* __restrict__ mask,
                            float* __restrict__ att)
{
  const int lane = threadIdx.x & 31, wv = threadIdx.x >> 5;
  const int tile = blockIdx.x * 4 + wv;           // 12 * 32 * 32 tiles
  const int h = tile >> 10, ti = (tile >> 5) & 31, tj = tile & 31;
  const int half = lane >> 4, mr = lane & 15;
  const int i0 = ti << 4, j0 = tj << 4;
  v16h a = ld_a16(Qa + ((size_t)h * Nn + i0 + mr) * 32, half);
  v16h b = ld_b16(Ka + ((size_t)h * Nn + j0 + mr) * 32, half);
  v8f acc = {};
  acc = __builtin_amdgcn_wmma_f32_16x16x32_f16(false, a, false, b, (short)0, acc, false, false);
  const int j = j0 + mr;
  const float mj = mask[j];
#pragma unroll
  for (int r = 0; r < 8; ++r) {
    int i = i0 + r + 8 * half;
    float v = acc[r] + z[((size_t)i * Nn + j) * Hh + h] + (mask[i] * mj - 1.f) * 1e9f;
    att[((size_t)h * Nn + i) * Nn + j] = v;
  }
}

// ---------------- row softmax -> f16 probabilities ------------------------------------
__global__ void softmax_rows(const float* __restrict__ att, _Float16* __restrict__ ah)
{
  __shared__ float red[128];
  const int row = blockIdx.x, t = threadIdx.x;
  const float* x = att + (size_t)row * Nn;
  _Float16* o = ah + (size_t)row * Nn;
  float m = -1e30f;
  for (int j = t; j < Nn; j += 128) m = fmaxf(m, x[j]);
  red[t] = m; __syncthreads();
  for (int s = 64; s > 0; s >>= 1) { if (t < s) red[t] = fmaxf(red[t], red[t + s]); __syncthreads(); }
  m = red[0]; __syncthreads();
  float e[4], sum = 0.f;
  for (int j = t, c = 0; j < Nn; j += 128, ++c) { e[c] = __expf(x[j] - m); sum += e[c]; }
  red[t] = sum; __syncthreads();
  for (int s = 64; s > 0; s >>= 1) { if (t < s) red[t] += red[t + s]; __syncthreads(); }
  const float inv = 1.f / red[0];
  for (int j = t, c = 0; j < Nn; j += 128, ++c) o[j] = (_Float16)(e[c] * inv);
}

// ---------------- pack [v | v_pts] as Bt (per head, N=48 padded) ----------------------
__global__ void build_btv(const float* __restrict__ proj, const float* __restrict__ vpts,
                          _Float16* __restrict__ btv)
{
  int t = blockIdx.x * blockDim.x + threadIdx.x;
  if (t >= Hh * 48 * Nn) return;
  int j = t & 511, n = (t >> 9) % 48, h = t / (48 * 512);
  float v;
  if (n < 16)      v = proj[(size_t)j * CTOT + 192 + h * 32 + 16 + n];
  else if (n < 40) v = vpts[(size_t)j * 288 + h * 24 + (n - 16)];
  else             v = 0.f;
  btv[((size_t)h * 48 + n) * Nn + j] = (_Float16)v;
}

// ---------------- o / o_pt: per-head A[512x512] x B[512x48] ---------------------------
__global__ void ov_gemm(const _Float16* __restrict__ ah, const _Float16* __restrict__ btv,
                        float* __restrict__ feat, float* __restrict__ optraw)
{
  const int lane = threadIdx.x & 31, wv = threadIdx.x >> 5;
  const int tile = blockIdx.x * 4 + wv;           // 12 * 32 * 3
  if (tile >= Hh * 32 * 3) return;
  const int h = tile / 96, ti = (tile % 96) / 3, tn = tile % 3;
  const int half = lane >> 4, mr = lane & 15;
  const int i0 = ti << 4, n0 = tn << 4;
  const _Float16* arow = ah + ((size_t)h * Nn + i0 + mr) * Nn;
  const _Float16* brow = btv + ((size_t)h * 48 + n0 + mr) * Nn;
  v8f acc = {};
  for (int k0 = 0; k0 < Nn; k0 += 32) {
    v16h a = ld_a16(arow + k0, half);
    v16h b = ld_b16(brow + k0, half);
    acc = __builtin_amdgcn_wmma_f32_16x16x32_f16(false, a, false, b, (short)0, acc, false, false);
  }
  const int n = n0 + mr;
#pragma unroll
  for (int r = 0; r < 8; ++r) {
    int i = i0 + r + 8 * half;
    if (n < 16) feat[(size_t)i * FEAT + h * C_Hd + n] = acc[r];
    else        optraw[((size_t)i * Hh + h) * 24 + (n - 16)] = acc[r];
  }
}

// ---------------- o_pair: per-i batched [12x512] x p[i][512x128] ----------------------
// p tile (32x128 f32, 16 KB) staged into LDS per K-step via ASYNCcnt-tracked
// global->LDS async copies (s_wait_asynccnt) where available, else vector load/store.
// WMMA B-fragment gathered from LDS (ds_load), A-fragment from attention probs.
__global__ void opair_gemm(const _Float16* __restrict__ ah, const float* __restrict__ p,
                           float* __restrict__ feat)
{
  __shared__ __align__(16) float ptile[32 * C_Pd];   // 16 KB
  const int i = blockIdx.x;
  const int tid = threadIdx.x;
  const int lane = tid & 31, wv = tid >> 5;          // 8 waves = 8 c-tiles
  const int half = lane >> 4, mr = lane & 15;
  const int c = (wv << 4) + mr;
  const float* pb = p + (size_t)i * Nn * C_Pd;
  v8f acc = {};
  for (int k0 = 0; k0 < Nn; k0 += 32) {
    const float* src = pb + (size_t)k0 * C_Pd;
#if USE_ASYNC_LDS
#pragma unroll
    for (int q = 0; q < 4; ++q) {
      int f4 = (tid + 256 * q) * 4;                  // float index, 16B granules
      __builtin_amdgcn_global_load_async_to_lds_b128(
          (AS1 i32x4*)(src + f4), (AS3 i32x4*)(ptile + f4), 0, 0);
    }
    __builtin_amdgcn_s_wait_asynccnt(0);
    __syncthreads();
#else
#pragma unroll
    for (int q = 0; q < 4; ++q) {
      int f4 = (tid + 256 * q) * 4;
      *(float4*)(ptile + f4) = *(const float4*)(src + f4);
    }
    __syncthreads();
#endif
    v16h a = {};
    if (mr < Hh) a = ld_a16(ah + ((size_t)mr * Nn + i) * Nn + k0, half);
    v16h b;
#pragma unroll
    for (int j = 0; j < 16; ++j) b[j] = (_Float16)ptile[(16 * half + j) * C_Pd + c];
    acc = __builtin_amdgcn_wmma_f32_16x16x32_f16(false, a, false, b, (short)0, acc, false, false);
    __syncthreads();
  }
#pragma unroll
  for (int r = 0; r < 8; ++r) {
    int h = r + 8 * half;
    if (h < Hh) feat[(size_t)i * FEAT + 576 + h * C_Pd + c] = acc[r];
  }
}

// ---------------- inverse-transform o_pt, compute norms -------------------------------
__global__ void opt_transform(const float* __restrict__ optraw, const float* __restrict__ rot,
                              const float* __restrict__ trans, float* __restrict__ feat)
{
  int t = blockIdx.x * blockDim.x + threadIdx.x;
  if (t >= Nn * Hh * 8) return;
  int i = t / 96, rem = t % 96, h = rem / 8, pp = rem % 8;
  const float* g = optraw + ((size_t)i * Hh + h) * 24 + pp * 3;
  const float* R = rot + i * 9; const float* tr = trans + i * 3;
  float dx = g[0] - tr[0], dy = g[1] - tr[1], dz = g[2] - tr[2];
  float lx = R[0] * dx + R[3] * dy + R[6] * dz;   // R^T (g - t)
  float ly = R[1] * dx + R[4] * dy + R[7] * dz;
  float lz = R[2] * dx + R[5] * dy + R[8] * dz;
  float* f = feat + (size_t)i * FEAT;
  f[192 + h * 24 + pp * 3 + 0] = lx;
  f[192 + h * 24 + pp * 3 + 1] = ly;
  f[192 + h * 24 + pp * 3 + 2] = lz;
  f[480 + h * 8 + pp] = sqrtf(lx * lx + ly * ly + lz * lz + 1e-8f);
}

// ---------------- utilities -----------------------------------------------------------
__global__ void cvt_f16(const float* __restrict__ s, _Float16* __restrict__ d, int n) {
  int t = blockIdx.x * blockDim.x + threadIdx.x;
  if (t < n) d[t] = (_Float16)s[t];
}
__global__ void transpose_f16(const float* __restrict__ W, _Float16* __restrict__ Wt,
                              int K, int Nsrc, int Npad) {
  int t = blockIdx.x * blockDim.x + threadIdx.x;
  if (t >= Npad * K) return;
  int n = t / K, k = t % K;
  Wt[(size_t)n * K + k] = (_Float16)(n < Nsrc ? W[(size_t)k * Nsrc + n] : 0.f);
}
__global__ void layernorm(const float* __restrict__ x, const float* __restrict__ g,
                          const float* __restrict__ b, float* __restrict__ y,
                          _Float16* __restrict__ yh) {
  __shared__ float red[128];
  const int i = blockIdx.x, t = threadIdx.x;
  const float* xr = x + (size_t)i * C_Sd;
  float s = 0.f;
  for (int c = t; c < C_Sd; c += 128) s += xr[c];
  red[t] = s; __syncthreads();
  for (int st = 64; st > 0; st >>= 1) { if (t < st) red[t] += red[t + st]; __syncthreads(); }
  const float mu = red[0] / C_Sd; __syncthreads();
  float v = 0.f;
  for (int c = t; c < C_Sd; c += 128) { float d = xr[c] - mu; v += d * d; }
  red[t] = v; __syncthreads();
  for (int st = 64; st > 0; st >>= 1) { if (t < st) red[t] += red[t + st]; __syncthreads(); }
  const float inv = rsqrtf(red[0] / C_Sd + 1e-5f);
  for (int c = t; c < C_Sd; c += 128) {
    float o = (xr[c] - mu) * inv * g[c] + b[c];
    y[(size_t)i * C_Sd + c] = o;
    if (yh) yh[(size_t)i * C_Sd + c] = (_Float16)o;
  }
}
__global__ void bb_gemv(const float* __restrict__ s, const float* __restrict__ w,
                        const float* __restrict__ b, float* __restrict__ upd) {
  int t = blockIdx.x * blockDim.x + threadIdx.x;
  if (t >= Nn * 6) return;
  int i = t / 6, c = t % 6;
  const float* sr = s + (size_t)i * C_Sd;
  float acc = b[c];
  for (int k = 0; k < C_Sd; ++k) acc += sr[k] * w[k * 6 + c];
  upd[t] = acc;
}
__global__ void frame_update(const float* __restrict__ upd, const float* __restrict__ rot,
                             const float* __restrict__ trans, float* __restrict__ rot_out,
                             float* __restrict__ trans_out) {
  int i = blockIdx.x * blockDim.x + threadIdx.x;
  if (i >= Nn) return;
  const float* u = upd + i * 6;
  float w = 1.f, x = u[0], y = u[1], z = u[2];
  float inv = rsqrtf(w * w + x * x + y * y + z * z);
  w *= inv; x *= inv; y *= inv; z *= inv;
  float Ru[9] = {
    1 - 2 * (y * y + z * z), 2 * (x * y - w * z), 2 * (x * z + w * y),
    2 * (x * y + w * z), 1 - 2 * (x * x + z * z), 2 * (y * z - w * x),
    2 * (x * z - w * y), 2 * (y * z + w * x), 1 - 2 * (x * x + y * y) };
  const float* R = rot + i * 9;
  float* Ro = rot_out + i * 9;
  for (int r = 0; r < 3; ++r)
    for (int c2 = 0; c2 < 3; ++c2)
      Ro[r * 3 + c2] = R[r * 3] * Ru[c2] + R[r * 3 + 1] * Ru[3 + c2] + R[r * 3 + 2] * Ru[6 + c2];
  const float* t = trans + i * 3;
  float* To = trans_out + i * 3;
  for (int r = 0; r < 3; ++r)
    To[r] = R[r * 3] * u[3] + R[r * 3 + 1] * u[4] + R[r * 3 + 2] * u[5] + t[r];
}

// =====================================================================================
extern "C" void kernel_launch(void* const* d_in, const int* in_sizes, int n_in,
                              void* d_out, int out_size, void* d_ws, size_t ws_size,
                              hipStream_t stream)
{
  (void)in_sizes; (void)n_in; (void)out_size; (void)ws_size;
  const float* s      = (const float*)d_in[0];
  const float* p      = (const float*)d_in[1];
  const float* rot    = (const float*)d_in[2];
  const float* trans  = (const float*)d_in[3];
  const float* mask   = (const float*)d_in[4];
  const float* q_w    = (const float*)d_in[5];
  const float* q_b    = (const float*)d_in[6];
  const float* kv_w   = (const float*)d_in[7];
  const float* kv_b   = (const float*)d_in[8];
  const float* qp_w   = (const float*)d_in[9];
  const float* qp_b   = (const float*)d_in[10];
  const float* kp_w   = (const float*)d_in[11];
  const float* kp_b   = (const float*)d_in[12];
  const float* vp_w   = (const float*)d_in[13];
  const float* vp_b   = (const float*)d_in[14];
  const float* b_w    = (const float*)d_in[15];
  const float* b_b    = (const float*)d_in[16];
  const float* head_w = (const float*)d_in[17];
  const float* out_w  = (const float*)d_in[18];
  const float* out_b  = (const float*)d_in[19];
  const float* ln1_g  = (const float*)d_in[20];
  const float* ln1_b  = (const float*)d_in[21];
  const float* t1_w   = (const float*)d_in[22];
  const float* t1_b   = (const float*)d_in[23];
  const float* t2_w   = (const float*)d_in[24];
  const float* t2_b   = (const float*)d_in[25];
  const float* t3_w   = (const float*)d_in[26];
  const float* t3_b   = (const float*)d_in[27];
  const float* ln2_g  = (const float*)d_in[28];
  const float* ln2_b  = (const float*)d_in[29];
  const float* bb_w   = (const float*)d_in[30];
  const float* bb_b   = (const float*)d_in[31];

  char* base = (char*)d_ws; size_t off = 0;
  auto alloc = [&](size_t bytes) -> void* {
    off = (off + 255) & ~(size_t)255;
    void* r = base + off; off += bytes; return r;
  };
  _Float16* sh     = (_Float16*)alloc((size_t)Nn * C_Sd * 2);
  _Float16* Wt     = (_Float16*)alloc((size_t)CTOT * C_Sd * 2);
  float*    ball   = (float*)   alloc((size_t)CTOT * 4);
  float*    proj   = (float*)   alloc((size_t)Nn * CTOT * 4);
  _Float16* Btb    = (_Float16*)alloc((size_t)16 * C_Pd * 2);
  float*    z      = (float*)   alloc((size_t)Nn * Nn * Hh * 4);
  float*    qpts   = (float*)   alloc((size_t)Nn * 144 * 4);
  float*    kpts   = (float*)   alloc((size_t)Nn * 144 * 4);
  float*    vpts   = (float*)   alloc((size_t)Nn * 288 * 4);
  _Float16* Qa     = (_Float16*)alloc((size_t)Hh * Nn * 32 * 2);
  _Float16* Ka     = (_Float16*)alloc((size_t)Hh * Nn * 32 * 2);
  float*    att    = (float*)   alloc((size_t)Hh * Nn * Nn * 4);
  _Float16* ahp    = (_Float16*)alloc((size_t)Hh * Nn * Nn * 2);
  _Float16* btv    = (_Float16*)alloc((size_t)Hh * 48 * Nn * 2);
  float*    optraw = (float*)   alloc((size_t)Nn * Hh * 24 * 4);
  float*    feat   = (float*)   alloc((size_t)Nn * FEAT * 4);
  _Float16* feath  = (_Float16*)alloc((size_t)Nn * FEAT * 2);
  _Float16* Wot    = (_Float16*)alloc((size_t)C_Sd * FEAT * 2);
  float*    sres   = (float*)   alloc((size_t)Nn * C_Sd * 4);
  float*    s1     = (float*)   alloc((size_t)Nn * C_Sd * 4);
  _Float16* s1h    = (_Float16*)alloc((size_t)Nn * C_Sd * 2);
  _Float16* t1t    = (_Float16*)alloc((size_t)C_Sd * C_Sd * 2);
  _Float16* t2t    = (_Float16*)alloc((size_t)C_Sd * C_Sd * 2);
  _Float16* t3t    = (_Float16*)alloc((size_t)C_Sd * C_Sd * 2);
  float*    h1     = (float*)   alloc((size_t)Nn * C_Sd * 4);
  _Float16* h1h    = (_Float16*)alloc((size_t)Nn * C_Sd * 2);
  float*    h2     = (float*)   alloc((size_t)Nn * C_Sd * 4);
  _Float16* h2h    = (_Float16*)alloc((size_t)Nn * C_Sd * 2);
  float*    s2     = (float*)   alloc((size_t)Nn * C_Sd * 4);
  float*    updv   = (float*)   alloc((size_t)Nn * 6 * 4);

  float* s_out     = (float*)d_out;
  float* rot_out   = s_out + (size_t)Nn * C_Sd;
  float* trans_out = rot_out + (size_t)Nn * 9;

  const dim3 b256(256);
  // --- prep: f16 activations + transposed f16 weights -----------------------------
  cvt_f16<<<(Nn * C_Sd + 255) / 256, b256, 0, stream>>>(s, sh, Nn * C_Sd);
  transpose_f16<<<(192 * 384 + 255) / 256, b256, 0, stream>>>(q_w,  Wt,                       384, 192, 192);
  transpose_f16<<<(384 * 384 + 255) / 256, b256, 0, stream>>>(kv_w, Wt + (size_t)192 * 384,   384, 384, 384);
  transpose_f16<<<(144 * 384 + 255) / 256, b256, 0, stream>>>(qp_w, Wt + (size_t)576 * 384,   384, 144, 144);
  transpose_f16<<<(144 * 384 + 255) / 256, b256, 0, stream>>>(kp_w, Wt + (size_t)720 * 384,   384, 144, 144);
  transpose_f16<<<(288 * 384 + 255) / 256, b256, 0, stream>>>(vp_w, Wt + (size_t)864 * 384,   384, 288, 288);
  transpose_f16<<<(16 * 128 + 255) / 256, b256, 0, stream>>>(b_w, Btb, 128, 12, 16);
  transpose_f16<<<(384 * 2112 + 255) / 256, b256, 0, stream>>>(out_w, Wot, 2112, 384, 384);
  transpose_f16<<<(384 * 384 + 255) / 256, b256, 0, stream>>>(t1_w, t1t, 384, 384, 384);
  transpose_f16<<<(384 * 384 + 255) / 256, b256, 0, stream>>>(t2_w, t2t, 384, 384, 384);
  transpose_f16<<<(384 * 384 + 255) / 256, b256, 0, stream>>>(t3_w, t3t, 384, 384, 384);
  (void)hipMemcpyAsync(ball,        q_b,  192 * 4, hipMemcpyDeviceToDevice, stream);
  (void)hipMemcpyAsync(ball + 192,  kv_b, 384 * 4, hipMemcpyDeviceToDevice, stream);
  (void)hipMemcpyAsync(ball + 576,  qp_b, 144 * 4, hipMemcpyDeviceToDevice, stream);
  (void)hipMemcpyAsync(ball + 720,  kp_b, 144 * 4, hipMemcpyDeviceToDevice, stream);
  (void)hipMemcpyAsync(ball + 864,  vp_b, 288 * 4, hipMemcpyDeviceToDevice, stream);

  // --- fused input projections: [512x384] x [384x1152] ----------------------------
  gemm_wmma<<<(32 * 72) / 4, 128, 0, stream>>>(sh, Wt, ball, nullptr, proj, nullptr, Nn, CTOT, C_Sd, 0);

  // --- points, augmented Q/K, pair bias, logits, softmax --------------------------
  to_pts<<<(Nn * 48 + 255) / 256, b256, 0, stream>>>(proj, 576, 48, rot, trans, qpts);
  to_pts<<<(Nn * 48 + 255) / 256, b256, 0, stream>>>(proj, 720, 48, rot, trans, kpts);
  to_pts<<<(Nn * 96 + 255) / 256, b256, 0, stream>>>(proj, 864, 96, rot, trans, vpts);
  prep_qk<<<(Hh * Nn + 255) / 256, b256, 0, stream>>>(proj, qpts, kpts, head_w, Qa, Ka);
  pair_bias_gemm<<<16384 / 4, 128, 0, stream>>>(p, Btb, b_b, z);
  attn_logits<<<(Hh * 32 * 32) / 4, 128, 0, stream>>>(Qa, Ka, z, mask, att);
  softmax_rows<<<Hh * Nn, 128, 0, stream>>>(att, ahp);

  // --- outputs of attention: o/o_pt and o_pair ------------------------------------
  build_btv<<<(Hh * 48 * Nn + 255) / 256, b256, 0, stream>>>(proj, vpts, btv);
  ov_gemm<<<(Hh * 32 * 3) / 4, 128, 0, stream>>>(ahp, btv, feat, optraw);
  opair_gemm<<<Nn, 256, 0, stream>>>(ahp, p, feat);
  opt_transform<<<(Nn * 96 + 255) / 256, b256, 0, stream>>>(optraw, rot, trans, feat);

  // --- output projection (+ residual s), LN, transition MLP, LN -------------------
  cvt_f16<<<(Nn * FEAT + 255) / 256, b256, 0, stream>>>(feat, feath, Nn * FEAT);
  gemm_wmma<<<(32 * 24) / 4, 128, 0, stream>>>(feath, Wot, out_b, s, sres, nullptr, Nn, C_Sd, FEAT, 0);
  layernorm<<<Nn, 128, 0, stream>>>(sres, ln1_g, ln1_b, s1, s1h);
  gemm_wmma<<<(32 * 24) / 4, 128, 0, stream>>>(s1h, t1t, t1_b, nullptr, h1, h1h, Nn, C_Sd, C_Sd, 1);
  gemm_wmma<<<(32 * 24) / 4, 128, 0, stream>>>(h1h, t2t, t2_b, nullptr, h2, h2h, Nn, C_Sd, C_Sd, 1);
  gemm_wmma<<<(32 * 24) / 4, 128, 0, stream>>>(h2h, t3t, t3_b, s1, s2, nullptr, Nn, C_Sd, C_Sd, 0);
  layernorm<<<Nn, 128, 0, stream>>>(s2, ln2_g, ln2_b, s_out, nullptr);

  // --- backbone update: quat compose + translate ----------------------------------
  bb_gemv<<<(Nn * 6 + 255) / 256, b256, 0, stream>>>(s_out, bb_w, bb_b, updv);
  frame_update<<<2, 256, 0, stream>>>(updv, rot, trans, rot_out, trans_out);
}
